// MultiHeadSelfAttention_35665408426215
// MI455X (gfx1250) — matrix operand order repaired and checked by reading, not compiled, not run on hardware
//
#include <hip/hip_runtime.h>
#include <hip/hip_bf16.h>

typedef __bf16 bf16_t;
typedef __attribute__((ext_vector_type(16))) __bf16 v16bf;
typedef __attribute__((ext_vector_type(8)))  __bf16 v8bf;
typedef __attribute__((ext_vector_type(4)))  __bf16 v4bf;
typedef __attribute__((ext_vector_type(8)))  float  v8f;

#define BD   2
#define SEQ  2048
#define DIN  1024
#define DOUT 1024
#define NH   16
#define HD   64

// ---------------------------------------------------------------------------
// WMMA helpers (CDNA5 gfx1250, wave32)
// ---------------------------------------------------------------------------
__device__ __forceinline__ v8f wmma_bf16(v16bf a, v16bf b, v8f c) {
  // D = A(16x32 bf16) * B(32x16 bf16) + C(16x16 f32)
  return __builtin_amdgcn_wmma_f32_16x16x32_bf16(false, a, false, b,
                                                 (short)0, c, false, false);
}

// Build a 16-element bf16 fragment from two 16-byte chunks.
__device__ __forceinline__ v16bf make_frag(const bf16_t* p0, const bf16_t* p1) {
  v8bf lo = *(const v8bf*)p0;
  v8bf hi = *(const v8bf*)p1;
  return __builtin_shufflevector(lo, hi, 0,1,2,3,4,5,6,7,8,9,10,11,12,13,14,15);
}

__device__ __forceinline__ float f4_get(const float4& v, int c) {
  switch (c) { case 0: return v.x; case 1: return v.y; case 2: return v.z; default: return v.w; }
}

// ---------------------------------------------------------------------------
// Kernel 1: fused QKV projection. C(4096x1024) = x(4096x1024) * W(1024x1024)
// blockIdx.z selects {Wq,Wk,Wv}. Q,K stored [b,h,s,d] bf16; V stored [b,h,d,s].
// 256 threads = 8 waves; 128x128 C tile; double-buffered LDS pipeline,
// one barrier per K-step; global prefetch overlaps the 8 WMMAs.
// B tile staged as 4x4 (k x n) blocks per thread: 4 coalesced b128 global
// loads -> 4 packed ds_store_b64 (k-contiguous per column).
// ---------------------------------------------------------------------------
__global__ __launch_bounds__(256) void qkv_gemm_kernel(
    const float* __restrict__ x,  const float* __restrict__ Wq,
    const float* __restrict__ Wk, const float* __restrict__ Wv,
    bf16_t* __restrict__ Qb, bf16_t* __restrict__ Kb, bf16_t* __restrict__ Vt)
{
  __shared__ __attribute__((aligned(16))) bf16_t As[2][128 * 32];  // [row][k]
  __shared__ __attribute__((aligned(16))) bf16_t Bs[2][128 * 32];  // [n][k]

  const int t     = threadIdx.x;
  const int ntile = blockIdx.x;   // 0..7
  const int mtile = blockIdx.y;   // 0..31
  const int which = blockIdx.z;   // 0=Q 1=K 2=V
  const float* W  = (which == 0) ? Wq : (which == 1) ? Wk : Wv;

  const int wave = t >> 5, lane = t & 31;
  const int hi = lane >> 4, ln = lane & 15;
  const int wm = wave & 3,  wn = wave >> 2;

  v8f acc[2][4];
#pragma unroll
  for (int mi = 0; mi < 2; ++mi)
#pragma unroll
    for (int ni = 0; ni < 4; ++ni)
      acc[mi][ni] = v8f{0.f,0.f,0.f,0.f,0.f,0.f,0.f,0.f};

  const int arow = t >> 1, akc = (t & 1) * 16;   // A: 2 thr/row, 16 f32 each
  const int bcol = (t & 31) * 4, bk0 = (t >> 5) * 4;  // B: 4x4 block per thread
  const float* aSrc = x + (size_t)(mtile * 128 + arow) * DIN + akc;
  const float* bSrc = W + (size_t)bk0 * DOUT + ntile * 128 + bcol;

  float4 aReg[4], bReg[4];
  auto prefetch = [&](int kt) {
    const float4* ap = (const float4*)(aSrc + kt * 32);
#pragma unroll
    for (int v = 0; v < 4; ++v) aReg[v] = ap[v];
#pragma unroll
    for (int j = 0; j < 4; ++j)
      bReg[j] = *(const float4*)(bSrc + (size_t)(kt * 32 + j) * DOUT);
  };
  auto stage = [&](int buf) {
#pragma unroll
    for (int v = 0; v < 4; ++v) {
      float4 f = aReg[v];
      bf16_t* d = &As[buf][arow * 32 + akc + v * 4];
      d[0] = (bf16_t)f.x; d[1] = (bf16_t)f.y;
      d[2] = (bf16_t)f.z; d[3] = (bf16_t)f.w;
    }
#pragma unroll
    for (int c = 0; c < 4; ++c) {              // column-packed b64 stores
      v4bf pk;
      pk[0] = (bf16_t)f4_get(bReg[0], c);
      pk[1] = (bf16_t)f4_get(bReg[1], c);
      pk[2] = (bf16_t)f4_get(bReg[2], c);
      pk[3] = (bf16_t)f4_get(bReg[3], c);
      *(v4bf*)&Bs[buf][(bcol + c) * 32 + bk0] = pk;
    }
  };

  prefetch(0);
  stage(0);
  __syncthreads();

  const int NT = DIN / 32;
  for (int kt = 0; kt < NT; ++kt) {
    const int cur = kt & 1;
    if (kt + 1 < NT) prefetch(kt + 1);          // global loads in flight

    v16bf afrag[2], bfrag[4];
#pragma unroll
    for (int mi = 0; mi < 2; ++mi) {
      const bf16_t* p = &As[cur][(wm * 32 + mi * 16 + ln) * 32 + 8 * hi];
      afrag[mi] = make_frag(p, p + 16);         // K = 8*hi+[0..7], 8*hi+16+[0..7]
    }
#pragma unroll
    for (int ni = 0; ni < 4; ++ni) {
      const bf16_t* p = &Bs[cur][(wn * 64 + ni * 16 + ln) * 32 + 8 * hi];
      bfrag[ni] = make_frag(p, p + 16);         // K = 8*hi+[0..7], 8*hi+16+[0..7]
    }
#pragma unroll
    for (int mi = 0; mi < 2; ++mi)
#pragma unroll
      for (int ni = 0; ni < 4; ++ni)
        acc[mi][ni] = wmma_bf16(afrag[mi], bfrag[ni], acc[mi][ni]);

    if (kt + 1 < NT) stage(cur ^ 1);            // fill the other buffer
    __syncthreads();                            // single barrier per K-step
  }

  // --- epilogue: uniform branch hoisted out of the store loops ---
  const int row0 = mtile * 128 + wm * 32;
  const int col0 = ntile * 128 + wn * 64 + ln;
  if (which == 2) {                             // V transposed [b,h,d,s]
#pragma unroll
    for (int ni = 0; ni < 4; ++ni) {
      const int gcol = col0 + ni * 16;
      const int h = gcol >> 6, d = gcol & (HD - 1);
#pragma unroll
      for (int mi = 0; mi < 2; ++mi)
#pragma unroll
        for (int r = 0; r < 8; ++r) {
          const int grow = row0 + mi * 16 + r + 8 * hi;
          const int b = grow >> 11, s = grow & (SEQ - 1);
          Vt[((size_t)(b * NH + h) * HD + d) * SEQ + s] = (bf16_t)acc[mi][ni][r];
        }
    }
  } else {                                      // Q/K: [b,h,s,d]
    bf16_t* __restrict__ dst = (which == 0) ? Qb : Kb;
#pragma unroll
    for (int ni = 0; ni < 4; ++ni) {
      const int gcol = col0 + ni * 16;
      const int h = gcol >> 6, d = gcol & (HD - 1);
#pragma unroll
      for (int mi = 0; mi < 2; ++mi)
#pragma unroll
        for (int r = 0; r < 8; ++r) {
          const int grow = row0 + mi * 16 + r + 8 * hi;
          const int b = grow >> 11, s = grow & (SEQ - 1);
          dst[((size_t)(b * NH + h) * SEQ + s) * HD + d] = (bf16_t)acc[mi][ni][r];
        }
    }
  }
}

// ---------------------------------------------------------------------------
// Kernel 2: causal flash attention.
// grid = (SEQ/128, B*H); 8 waves, each owns 16 query rows (per-wave key loop,
// no intra-loop barriers => true causal work halving).
// ---------------------------------------------------------------------------
__global__ __launch_bounds__(256) void flash_attn_kernel(
    const bf16_t* __restrict__ Qb, const bf16_t* __restrict__ Kb,
    const bf16_t* __restrict__ Vt, bf16_t* __restrict__ Ctx)
{
  __shared__ __attribute__((aligned(16))) bf16_t P[8][16 * 32];  // per-wave P tile

  const int t = threadIdx.x;
  const int wave = t >> 5, lane = t & 31, hi = lane >> 4, ln = lane & 15;
  const int qt = blockIdx.x, bh = blockIdx.y;
  const int qbase = qt * 128 + wave * 16;

  const bf16_t* Qp = Qb + (size_t)bh * SEQ * HD;
  const bf16_t* Kp = Kb + (size_t)bh * SEQ * HD;
  const bf16_t* Vp = Vt + (size_t)bh * HD * SEQ;

  // Q A-fragments (row = qbase+ln), two 32-wide K halves over HD=64
  v16bf qa[2];
#pragma unroll
  for (int f = 0; f < 2; ++f) {
    const bf16_t* p = Qp + (size_t)(qbase + ln) * HD + f * 32 + 16 * hi;
    qa[f] = make_frag(p, p + 8);
  }

  const v8f zero = v8f{0.f,0.f,0.f,0.f,0.f,0.f,0.f,0.f};
  v8f accv[4];
#pragma unroll
  for (int j = 0; j < 4; ++j) accv[j] = zero;
  float Mrow[8], Lrow[8];
#pragma unroll
  for (int r = 0; r < 8; ++r) { Mrow[r] = -3.0e38f; Lrow[r] = 0.f; }

  const int kend = qbase + 15;            // highest key this wave needs
  for (int kb = 0; kb <= kend; kb += 32) {
    // hint the next key/value tiles into cache while this one computes
    if (kb + 32 <= kend) {
      __builtin_prefetch(Kp + (size_t)(kb + 32 + ln) * HD, 0, 1);
      __builtin_prefetch(Vp + (size_t)(lane * 2) * SEQ + kb + 32, 0, 1);
    }

    // ---- scores: S(16q x 32k) as two 16x16 WMMA C-tiles ----
    const bf16_t* kp0 = Kp + (size_t)(kb + ln) * HD;        // keys kb..kb+15
    const bf16_t* kp1 = kp0 + 16 * HD;                      // keys kb+16..kb+31
    v16bf b00 = make_frag(kp0 + 16 * hi,      kp0 + 16 * hi + 8);       // d 0..31
    v16bf b01 = make_frag(kp0 + 32 + 16 * hi, kp0 + 32 + 16 * hi + 8);  // d 32..63
    v16bf b10 = make_frag(kp1 + 16 * hi,      kp1 + 16 * hi + 8);
    v16bf b11 = make_frag(kp1 + 32 + 16 * hi, kp1 + 32 + 16 * hi + 8);

    v8f s0 = wmma_bf16(qa[0], b00, zero); s0 = wmma_bf16(qa[1], b01, s0);
    v8f s1 = wmma_bf16(qa[0], b10, zero); s1 = wmma_bf16(qa[1], b11, s1);

    // ---- online softmax (rows split across 16-lane halves) ----
    float alph[8];
#pragma unroll
    for (int r = 0; r < 8; ++r) {
      const int qr = qbase + r + 8 * hi;
      float a0 = s0[r] * 0.125f; if (kb + ln      > qr) a0 = -3.0e38f;
      float a1 = s1[r] * 0.125f; if (kb + 16 + ln > qr) a1 = -3.0e38f;
      float m = fmaxf(a0, a1);
      m = fmaxf(m, __shfl_xor(m, 1, 16));
      m = fmaxf(m, __shfl_xor(m, 2, 16));
      m = fmaxf(m, __shfl_xor(m, 4, 16));
      m = fmaxf(m, __shfl_xor(m, 8, 16));
      const float mn = fmaxf(Mrow[r], m);
      const float al = __expf(Mrow[r] - mn);
      Mrow[r] = mn;
      const float p0 = __expf(a0 - mn);
      const float p1 = __expf(a1 - mn);
      float rs = p0 + p1;
      rs += __shfl_xor(rs, 1, 16);
      rs += __shfl_xor(rs, 2, 16);
      rs += __shfl_xor(rs, 4, 16);
      rs += __shfl_xor(rs, 8, 16);
      Lrow[r] = Lrow[r] * al + rs;
      alph[r] = al;
      // C-layout -> LDS row-major P[qrow][key]
      P[wave][(r + 8 * hi) * 32 + ln]      = (bf16_t)p0;
      P[wave][(r + 8 * hi) * 32 + 16 + ln] = (bf16_t)p1;
    }
    // make the wave's own DS stores visible before re-reading as A-fragment
    asm volatile("s_wait_dscnt 0x0" ::: "memory");

    const bf16_t* pp = &P[wave][ln * 32 + 16 * hi];
    v16bf pa = make_frag(pp, pp + 8);    // P as A(16x32) fragment

    // ---- ctx += P @ V : V^T columns are contiguous over keys ----
#pragma unroll
    for (int j = 0; j < 4; ++j) {
      const bf16_t* vp = Vp + (size_t)(j * 16 + ln) * SEQ + kb + 16 * hi;
      v16bf vb = make_frag(vp, vp + 8);
#pragma unroll
      for (int r = 0; r < 8; ++r) accv[j][r] *= alph[r];
      accv[j] = wmma_bf16(pa, vb, accv[j]);
    }
  }

  // ---- finalize: ctx[b][s][h*64+d] in bf16 for the O-projection ----
  const int b = bh >> 4, h = bh & 15;
#pragma unroll
  for (int r = 0; r < 8; ++r) {
    const float inv = 1.0f / Lrow[r];
    const int srow = qbase + r + 8 * hi;
#pragma unroll
    for (int j = 0; j < 4; ++j)
      Ctx[(size_t)(b * SEQ + srow) * DOUT + h * HD + j * 16 + ln] =
          (bf16_t)(accv[j][r] * inv);
  }
}

// ---------------------------------------------------------------------------
// Kernel 3: out = ctx(bf16, 4096x1024) @ W_o + b_o  -> f32
// Same double-buffered pipeline; A tile is already bf16.
// ---------------------------------------------------------------------------
__global__ __launch_bounds__(256) void o_gemm_kernel(
    const bf16_t* __restrict__ Ctx, const float* __restrict__ Wo,
    const float* __restrict__ bo, float* __restrict__ out)
{
  __shared__ __attribute__((aligned(16))) bf16_t As[2][128 * 32];
  __shared__ __attribute__((aligned(16))) bf16_t Bs[2][128 * 32];

  const int t = threadIdx.x;
  const int ntile = blockIdx.x, mtile = blockIdx.y;
  const int wave = t >> 5, lane = t & 31;
  const int hi = lane >> 4, ln = lane & 15;
  const int wm = wave & 3,  wn = wave >> 2;

  v8f acc[2][4];
#pragma unroll
  for (int mi = 0; mi < 2; ++mi)
#pragma unroll
    for (int ni = 0; ni < 4; ++ni)
      acc[mi][ni] = v8f{0.f,0.f,0.f,0.f,0.f,0.f,0.f,0.f};

  const int arow = t >> 1, akc = (t & 1) * 16;
  const int bcol = (t & 31) * 4, bk0 = (t >> 5) * 4;
  const bf16_t* aSrc = Ctx + (size_t)(mtile * 128 + arow) * DOUT + akc;
  const float*  bSrc = Wo  + (size_t)bk0 * DOUT + ntile * 128 + bcol;

  v8bf  aReg[2];
  float4 bReg[4];
  auto prefetch = [&](int kt) {
    const v8bf* ap = (const v8bf*)(aSrc + kt * 32);
    aReg[0] = ap[0]; aReg[1] = ap[1];
#pragma unroll
    for (int j = 0; j < 4; ++j)
      bReg[j] = *(const float4*)(bSrc + (size_t)(kt * 32 + j) * DOUT);
  };
  auto stage = [&](int buf) {
    *(v8bf*)&As[buf][arow * 32 + akc]     = aReg[0];
    *(v8bf*)&As[buf][arow * 32 + akc + 8] = aReg[1];
#pragma unroll
    for (int c = 0; c < 4; ++c) {
      v4bf pk;
      pk[0] = (bf16_t)f4_get(bReg[0], c);
      pk[1] = (bf16_t)f4_get(bReg[1], c);
      pk[2] = (bf16_t)f4_get(bReg[2], c);
      pk[3] = (bf16_t)f4_get(bReg[3], c);
      *(v4bf*)&Bs[buf][(bcol + c) * 32 + bk0] = pk;
    }
  };

  prefetch(0);
  stage(0);
  __syncthreads();

  const int NT = DOUT / 32;
  for (int kt = 0; kt < NT; ++kt) {
    const int cur = kt & 1;
    if (kt + 1 < NT) prefetch(kt + 1);

    v16bf afrag[2], bfrag[4];
#pragma unroll
    for (int mi = 0; mi < 2; ++mi) {
      const bf16_t* p = &As[cur][(wm * 32 + mi * 16 + ln) * 32 + 8 * hi];
      afrag[mi] = make_frag(p, p + 16);
    }
#pragma unroll
    for (int ni = 0; ni < 4; ++ni) {
      const bf16_t* p = &Bs[cur][(wn * 64 + ni * 16 + ln) * 32 + 8 * hi];
      bfrag[ni] = make_frag(p, p + 16);
    }
#pragma unroll
    for (int mi = 0; mi < 2; ++mi)
#pragma unroll
      for (int ni = 0; ni < 4; ++ni)
        acc[mi][ni] = wmma_bf16(afrag[mi], bfrag[ni], acc[mi][ni]);

    if (kt + 1 < NT) stage(cur ^ 1);
    __syncthreads();
  }

  const int row0 = mtile * 128 + wm * 32;
#pragma unroll
  for (int ni = 0; ni < 4; ++ni) {
    const int gcol = ntile * 128 + wn * 64 + ni * 16 + ln;
    const float bias = bo[gcol];
#pragma unroll
    for (int mi = 0; mi < 2; ++mi)
#pragma unroll
      for (int r = 0; r < 8; ++r) {
        const int grow = row0 + mi * 16 + r + 8 * hi;
        out[(size_t)grow * DOUT + gcol] = acc[mi][ni][r] + bias;
      }
  }
}

// ---------------------------------------------------------------------------
extern "C" void kernel_launch(void* const* d_in, const int* in_sizes, int n_in,
                              void* d_out, int out_size, void* d_ws, size_t ws_size,
                              hipStream_t stream) {
  (void)in_sizes; (void)n_in; (void)out_size; (void)ws_size;
  const float* x  = (const float*)d_in[0];
  const float* Wq = (const float*)d_in[1];
  const float* Wk = (const float*)d_in[2];
  const float* Wv = (const float*)d_in[3];
  const float* Wo = (const float*)d_in[4];
  const float* bo = (const float*)d_in[5];
  float* out = (float*)d_out;

  char* ws = (char*)d_ws;
  const size_t SZ = (size_t)BD * NH * SEQ * HD * sizeof(bf16_t);  // 8 MB each
  bf16_t* Qb  = (bf16_t*)(ws);
  bf16_t* Kb  = (bf16_t*)(ws + SZ);
  bf16_t* Vt  = (bf16_t*)(ws + 2 * SZ);
  bf16_t* Ctx = (bf16_t*)(ws + 3 * SZ);

  qkv_gemm_kernel<<<dim3(DOUT / 128, (BD * SEQ) / 128, 3), 256, 0, stream>>>(
      x, Wq, Wk, Wv, Qb, Kb, Vt);
  flash_attn_kernel<<<dim3(SEQ / 128, BD * NH), 256, 0, stream>>>(Qb, Kb, Vt, Ctx);
  o_gemm_kernel<<<dim3(DOUT / 128, (BD * SEQ) / 128), 256, 0, stream>>>(
      Ctx, Wo, bo, out);
}
